// UPRL_LM_17824114278733
// MI455X (gfx1250) — matrix operand order repaired and checked by reading, not compiled, not run on hardware
//
#include <hip/hip_runtime.h>
#include <math.h>

// ---------------------------------------------------------------------------
// CDNA5 (gfx1250, wave32) implementation of the UPRL policy loop.
// Bulk FLOPs (6.5 TFLOP of LSTM gate GEMMs) run through v_wmma_f32_16x16x32_f16
// with f16 inputs / f32 accumulation. Recurrence is chained per timestep:
//   gates = x_t @ Wih^T + b          (WMMA GEMM, bias epilogue)
//   gates += h  @ Whh^T              (WMMA GEMM, C-add epilogue)
//   (h,c)  = lstm_cell(gates, c)     (VALU)
// Scoring head is algebraically reduced (one-hot block of emb2 selects a
// column of u = mlp1_W^T rep100) and done in VALU (<0.01% of FLOPs).
// ---------------------------------------------------------------------------

typedef _Float16 h16 __attribute__((ext_vector_type(16)));
typedef _Float16 h8  __attribute__((ext_vector_type(8)));
typedef float    f8  __attribute__((ext_vector_type(8)));

#define B_Q      4096
#define L_Q      15
#define NINP_Q   256
#define NHID_Q   512
#define GATES_Q  2048      // 4*NHID
#define K_Q      42
#define NSPLIT_Q 20
#define PAD_Q    30003
#define IN0P_Q   544       // 513 padded to mult of 32
#define IN1_Q    1024
#define REPP_Q   1056      // 1040 padded to mult of 32

__device__ __forceinline__ float sigf(float x) { return 1.0f / (1.0f + __expf(-x)); }

// ---------------------------------------------------------------------------
// WMMA GEMM:  Cout[M,N] = A[M,Kp] @ W[N,Kp]^T  (+ bias[N] | + Cin[M,N])
// A, W in f16 (row-major, Kp multiple of 32), Cout f32 row-major (ld = N).
// cmode: 1 = add bias[N], 2 = add Cin[M,N] (may alias Cout), else none.
// Block: 256 thr (8 wave32), tile 128x128; wave tile 64x32 = 4x2 WMMA tiles.
// ---------------------------------------------------------------------------
__global__ __launch_bounds__(256)
void gemm_wmma_f16(const _Float16* __restrict__ A, const _Float16* __restrict__ W,
                   const float* __restrict__ bias, const float* __restrict__ Cin,
                   float* __restrict__ Cout, int M, int N, int Kp, int cmode)
{
    __shared__ _Float16 As[128 * 32] __attribute__((aligned(32)));
    __shared__ _Float16 Bs[128 * 32] __attribute__((aligned(32)));

    const int tid   = threadIdx.x;
    const int lane  = tid & 31;
    const int wid   = tid >> 5;       // 0..7
    const int wm    = wid >> 2;       // 0..1 : 64-row band
    const int wn    = wid & 3;        // 0..3 : 32-col band
    const int bm    = blockIdx.y * 128;
    const int bn    = blockIdx.x * 128;
    const int l16   = lane & 15;
    const int khalf = lane >> 4;

    f8 acc[4][2];
#pragma unroll
    for (int i = 0; i < 4; i++)
#pragma unroll
        for (int j = 0; j < 2; j++)
#pragma unroll
            for (int v = 0; v < 8; v++) acc[i][j][v] = 0.0f;

    const int lrow = tid >> 1;          // 0..127
    const int lcol = (tid & 1) << 4;    // 0 or 16 (halves)

    for (int k0 = 0; k0 < Kp; k0 += 32) {
        h16 av, bv;
#pragma unroll
        for (int e = 0; e < 16; e++) { av[e] = (_Float16)0.0f; bv[e] = (_Float16)0.0f; }
        const int gm = bm + lrow;
        if (gm < M) av = *(const h16*)(A + (size_t)gm * Kp + k0 + lcol);
        const int gn = bn + lrow;
        if (gn < N) bv = *(const h16*)(W + (size_t)gn * Kp + k0 + lcol);
        *(h16*)(As + lrow * 32 + lcol) = av;
        *(h16*)(Bs + lrow * 32 + lcol) = bv;
        __syncthreads();

        // B fragment: lane holds column (wn*32 + j*16 + l16), K = khalf*8..+7 and 16+khalf*8..+7
        h16 bf[2];
#pragma unroll
        for (int j = 0; j < 2; j++) {
            const int col = wn * 32 + j * 16 + l16;
            const _Float16* bp = Bs + col * 32 + khalf * 8;
            h8 blo = *(const h8*)(bp);
            h8 bhi = *(const h8*)(bp + 16);
#pragma unroll
            for (int e = 0; e < 8; e++) { bf[j][e] = blo[e]; bf[j][e + 8] = bhi[e]; }
        }
        // A fragment: lane holds row, K = khalf*8..+7 and 16+khalf*8..+7
#pragma unroll
        for (int i = 0; i < 4; i++) {
            const int row = wm * 64 + i * 16 + l16;
            const _Float16* ap = As + row * 32 + khalf * 8;
            h8 lo = *(const h8*)(ap);
            h8 hi = *(const h8*)(ap + 16);
            h16 af;
#pragma unroll
            for (int e = 0; e < 8; e++) { af[e] = lo[e]; af[e + 8] = hi[e]; }
#pragma unroll
            for (int j = 0; j < 2; j++) {
                acc[i][j] = __builtin_amdgcn_wmma_f32_16x16x32_f16(
                    false, af, false, bf[j], (short)0, acc[i][j], false, false);
            }
        }
        __syncthreads();
    }

    // Epilogue: lane holds C rows (lane>>4)*8 + v, col lane&15 within each tile
    const int lh = lane >> 4;
#pragma unroll
    for (int i = 0; i < 4; i++) {
#pragma unroll
        for (int j = 0; j < 2; j++) {
            const int gn = bn + wn * 32 + j * 16 + l16;
            if (gn >= N) continue;
            const float badd = (cmode == 1) ? bias[gn] : 0.0f;
            const int gm0 = bm + wm * 64 + i * 16 + lh * 8;
#pragma unroll
            for (int v = 0; v < 8; v++) {
                const int gm = gm0 + v;
                if (gm < M) {
                    float x = acc[i][j][v] + badd;
                    if (cmode == 2) x += Cin[(size_t)gm * N + gn];
                    Cout[(size_t)gm * N + gn] = x;
                }
            }
        }
    }
}

// ------------------------- support kernels ---------------------------------

__global__ void k_init_state(const int* __restrict__ ids, const int* __restrict__ slen,
                             int* __restrict__ st, int* __restrict__ len, int n)
{
    int i = blockIdx.x * blockDim.x + threadIdx.x;
    if (i < n) st[i] = ids[i];
    if (i < B_Q) len[i] = slen[i];
}

__global__ void k_f32_to_f16_pad(const float* __restrict__ src, _Float16* __restrict__ dst,
                                 int rows, int ksrc, int kdst)
{
    long long i = (long long)blockIdx.x * blockDim.x + threadIdx.x;
    long long total = (long long)rows * kdst;
    if (i >= total) return;
    int c = (int)(i % kdst);
    long long r = i / kdst;
    float v = (c < ksrc) ? src[r * ksrc + c] : 0.0f;
    dst[i] = (_Float16)v;
}

// X0[t*B+b][c], c<256: emb[st]*(t!=pos); c<512: emb[ids0]; c==512: (t==pos); pad 0
__global__ void k_build_x(const int* __restrict__ st, const int* __restrict__ ids0,
                          const float* __restrict__ emb, _Float16* __restrict__ X0, int pos)
{
    int idx = blockIdx.x * blockDim.x + threadIdx.x;
    const int total = L_Q * B_Q * IN0P_Q;
    if (idx >= total) return;
    int c = idx % IN0P_Q;
    int row = idx / IN0P_Q;
    int t = row / B_Q;
    int b = row % B_Q;
    float v = 0.0f;
    if (c < NINP_Q) {
        int tok = st[b * L_Q + t];
        v = (t == pos) ? 0.0f : emb[(size_t)tok * NINP_Q + c];
    } else if (c < 2 * NINP_Q) {
        int tok = ids0[b * L_Q + t];
        v = emb[(size_t)tok * NINP_Q + (c - NINP_Q)];
    } else if (c == 2 * NINP_Q) {
        v = (t == pos) ? 1.0f : 0.0f;
    }
    X0[idx] = (_Float16)v;
}

// gates (B,2048) split i|f|g|o; update C, H(f16); optionally write h to dest
__global__ void k_lstm_cell(const float* __restrict__ gates, float* __restrict__ C,
                            _Float16* __restrict__ H, _Float16* __restrict__ dest,
                            int dest_ld, int dest_off, int do_dest)
{
    int i = blockIdx.x * blockDim.x + threadIdx.x;
    if (i >= B_Q * NHID_Q) return;
    int b = i >> 9, j = i & (NHID_Q - 1);
    const float* g = gates + (size_t)b * GATES_Q;
    float ig = sigf(g[j]);
    float fg = sigf(g[NHID_Q + j]);
    float gg = tanhf(g[2 * NHID_Q + j]);
    float og = sigf(g[3 * NHID_Q + j]);
    float c = fg * C[i] + ig * gg;
    float h = og * tanhf(c);
    C[i] = c;
    H[i] = (_Float16)h;
    if (do_dest) dest[(size_t)b * dest_ld + dest_off + j] = (_Float16)h;
}

// rep cols 1024..1055: key_pos scalar, 15-wide pos one-hot, zero pad
__global__ void k_rep_extra(const float* __restrict__ key_pos, _Float16* __restrict__ rep, int pos)
{
    int i = blockIdx.x * blockDim.x + threadIdx.x;
    if (i >= B_Q * 32) return;
    int b = i >> 5, j = i & 31;
    float v = 0.0f;
    if (j == 0) v = key_pos[b * L_Q + pos];
    else if (j <= L_Q) v = ((j - 1) == pos) ? 1.0f : 0.0f;
    rep[(size_t)b * REPP_Q + 1024 + j] = (_Float16)v;
}

// u[b][d] = sum_j rep100[b][j] * mlp1_W[j][d],  d < 298
__global__ void k_u(const float* __restrict__ rep100, const float* __restrict__ mlp1W,
                    float* __restrict__ u)
{
    int i = blockIdx.x * blockDim.x + threadIdx.x;
    if (i >= B_Q * 298) return;
    int b = i / 298, d = i % 298;
    const float* r = rep100 + (size_t)b * 100;
    float s = 0.0f;
    for (int j = 0; j < 100; j++) s += r[j] * mlp1W[j * 298 + d];
    u[i] = s;
}

// cb[b] = mlp1_b . rep100[b]
__global__ void k_cb(const float* __restrict__ rep100, const float* __restrict__ mlp1b,
                     float* __restrict__ cb)
{
    int b = blockIdx.x * blockDim.x + threadIdx.x;
    if (b >= B_Q) return;
    const float* r = rep100 + (size_t)b * 100;
    float s = 0.0f;
    for (int j = 0; j < 100; j++) s += mlp1b[j] * r[j];
    cb[b] = s;
}

// score[b][k] = enc(word_k).u[b][0:256] + u[b][256+k] + cb[b]
__global__ void k_score(const float* __restrict__ u, const float* __restrict__ cb,
                        const float* __restrict__ enc, const int* __restrict__ rep_words_all,
                        const float* __restrict__ delv, const float* __restrict__ holdv,
                        float* __restrict__ score, int step)
{
    int i = blockIdx.x * blockDim.x + threadIdx.x;
    if (i >= B_Q * K_Q) return;
    int b = i / K_Q, k = i % K_Q;
    const float* e;
    if (k < 2 * NSPLIT_Q) {
        int w = rep_words_all[((size_t)b * 5 + step) * 40 + k];
        e = enc + (size_t)w * NINP_Q;
    } else if (k == 2 * NSPLIT_Q) e = delv;
    else e = holdv;
    const float* ub = u + (size_t)b * 298;
    float s = 0.0f;
    for (int d = 0; d < NINP_Q; d++) s += e[d] * ub[d];
    s += ub[NINP_Q + k] + cb[b];
    score[i] = s;
}

// softmax/argmax, pi, flags, tail edit, outputs
__global__ void k_act(const float* __restrict__ score, const int* __restrict__ rep_words_all,
                      const int* __restrict__ valid_all, int* __restrict__ st, int* __restrict__ len,
                      float* __restrict__ out_pis, float* __restrict__ out_acts,
                      float* __restrict__ out_st, float* __restrict__ out_len,
                      int pos, int step)
{
    int b = blockIdx.x * blockDim.x + threadIdx.x;
    if (b >= B_Q) return;
    const float* s = score + (size_t)b * K_Q;
    float mx = s[0]; int a = 0;
    for (int k = 1; k < K_Q; k++) { float v = s[k]; if (v > mx) { mx = v; a = k; } }
    float sum = 0.0f;
    for (int k = 0; k < K_Q; k++) sum += __expf(s[k] - mx);
    float pi = __expf(s[a] - mx) / sum;

    int valid = valid_all[(size_t)b * 5 + step];
    int lt = len[b];
    int repf = (a < NSPLIT_Q) && valid;
    int insf = (a >= NSPLIT_Q) && (a < 2 * NSPLIT_Q) && valid && (lt + 1 <= 15);
    int delf = (a == 2 * NSPLIT_Q) && valid && (lt - 1 > 2);
    int awi = (a > NSPLIT_Q - 1) ? (NSPLIT_Q - 1) : a;
    int aw = rep_words_all[((size_t)b * 5 + step) * 40 + awi];

    int* stb = st + (size_t)b * L_Q;
    const int TL = L_Q - pos;
    int old[16], nw[16];
    for (int j = 0; j < TL; j++) old[j] = stb[pos + j];
    if (repf)      { nw[0] = aw; for (int j = 1; j < TL; j++) nw[j] = old[j]; }
    else if (insf) { nw[0] = aw; for (int j = 1; j < TL; j++) nw[j] = old[j - 1]; }
    else if (delf) { for (int j = 0; j < TL - 1; j++) nw[j] = old[j + 1]; nw[TL - 1] = PAD_Q; }
    else           { for (int j = 0; j < TL; j++) nw[j] = old[j]; }
    for (int j = 0; j < TL; j++) stb[pos + j] = nw[j];
    int lnew = lt + insf - delf;
    len[b] = lnew;

    out_pis[(size_t)b * 5 + step]  = pi;
    out_acts[(size_t)b * 5 + step] = (float)a;
    for (int j = 0; j < L_Q; j++) out_st[(size_t)b * L_Q + j] = (float)stb[j];
    out_len[b] = (float)lnew;
}

// --------------------------- orchestration ---------------------------------

extern "C" void kernel_launch(void* const* d_in, const int* in_sizes, int n_in,
                              void* d_out, int out_size, void* d_ws, size_t ws_size,
                              hipStream_t stream)
{
    const int*   ids    = (const int*)d_in[0];
    const float* keypos = (const float*)d_in[1];
    const int*   seqlen = (const int*)d_in[2];
    const int*   repw   = (const int*)d_in[3];
    const int*   valid  = (const int*)d_in[4];
    const float* embT   = (const float*)d_in[5];
    const float* encT   = (const float*)d_in[6];
    const float* Wih0   = (const float*)d_in[7];
    const float* Whh0   = (const float*)d_in[8];
    const float* b0     = (const float*)d_in[9];
    const float* Wih1   = (const float*)d_in[10];
    const float* Whh1   = (const float*)d_in[11];
    const float* b1     = (const float*)d_in[12];
    const float* delv   = (const float*)d_in[13];
    const float* holdv  = (const float*)d_in[14];
    const float* mlpW   = (const float*)d_in[15];
    const float* mlpb   = (const float*)d_in[16];
    const float* mlp1W  = (const float*)d_in[17];
    const float* mlp1b  = (const float*)d_in[18];
    (void)in_sizes; (void)n_in; (void)out_size; (void)ws_size;

    char* ws = (char*)d_ws;
    size_t off = 0;
    auto alloc = [&](size_t bytes) -> char* {
        char* p = ws + off;
        off += (bytes + 255) & ~(size_t)255;
        return p;
    };

    int*      st     = (int*)alloc((size_t)B_Q * L_Q * 4);
    int*      len    = (int*)alloc((size_t)B_Q * 4);
    _Float16* X0     = (_Float16*)alloc((size_t)L_Q * B_Q * IN0P_Q * 2);
    _Float16* X1     = (_Float16*)alloc((size_t)L_Q * B_Q * IN1_Q * 2);
    _Float16* Wih0h  = (_Float16*)alloc((size_t)2 * GATES_Q * IN0P_Q * 2);
    _Float16* Whh0h  = (_Float16*)alloc((size_t)2 * GATES_Q * NHID_Q * 2);
    _Float16* Wih1h  = (_Float16*)alloc((size_t)2 * GATES_Q * IN1_Q * 2);
    _Float16* Whh1h  = (_Float16*)alloc((size_t)2 * GATES_Q * NHID_Q * 2);
    _Float16* mlpWh  = (_Float16*)alloc((size_t)100 * REPP_Q * 2);
    float*    gates  = (float*)alloc((size_t)B_Q * GATES_Q * 4);
    _Float16* Hf16   = (_Float16*)alloc((size_t)B_Q * NHID_Q * 2);
    float*    Cbuf   = (float*)alloc((size_t)B_Q * NHID_Q * 4);
    _Float16* repb   = (_Float16*)alloc((size_t)B_Q * REPP_Q * 2);
    float*    rep100 = (float*)alloc((size_t)B_Q * 100 * 4);
    float*    ubuf   = (float*)alloc((size_t)B_Q * 298 * 4);
    float*    cbb    = (float*)alloc((size_t)B_Q * 4);
    float*    scoreb = (float*)alloc((size_t)B_Q * K_Q * 4);

    float* out_pis = (float*)d_out;
    float* out_act = out_pis + (size_t)B_Q * 5;
    float* out_st  = out_act + (size_t)B_Q * 5;
    float* out_len = out_st + (size_t)B_Q * L_Q;

    k_init_state<<<(B_Q * L_Q + 255) / 256, 256, 0, stream>>>(ids, seqlen, st, len, B_Q * L_Q);

    auto conv = [&](const float* src, _Float16* dst, int rows, int ksrc, int kdst) {
        long long total = (long long)rows * kdst;
        k_f32_to_f16_pad<<<(int)((total + 255) / 256), 256, 0, stream>>>(src, dst, rows, ksrc, kdst);
    };
    conv(Wih0, Wih0h, 2 * GATES_Q, 513, IN0P_Q);
    conv(Whh0, Whh0h, 2 * GATES_Q, NHID_Q, NHID_Q);
    conv(Wih1, Wih1h, 2 * GATES_Q, IN1_Q, IN1_Q);
    conv(Whh1, Whh1h, 2 * GATES_Q, NHID_Q, NHID_Q);
    conv(mlpW, mlpWh, 100, 1040, REPP_Q);

    dim3 gemm_grid(GATES_Q / 128, B_Q / 128);   // 16 x 32

    for (int step = 0; step < 5; step++) {
        int pos = step % (L_Q - 3) + 2;

        {
            int total = L_Q * B_Q * IN0P_Q;
            k_build_x<<<(total + 255) / 256, 256, 0, stream>>>(st, ids, embT, X0, pos);
        }

        for (int layer = 0; layer < 2; layer++) {
            const _Float16* Xin  = (layer == 0) ? X0 : X1;
            const int       xld  = (layer == 0) ? IN0P_Q : IN1_Q;
            const _Float16* WihH = (layer == 0) ? Wih0h : Wih1h;
            const _Float16* WhhH = (layer == 0) ? Whh0h : Whh1h;
            const float*    bb   = (layer == 0) ? b0 : b1;

            for (int dir = 0; dir < 2; dir++) {     // 0 = fwd, 1 = reverse
                hipMemsetAsync(Hf16, 0, (size_t)B_Q * NHID_Q * 2, stream);
                hipMemsetAsync(Cbuf, 0, (size_t)B_Q * NHID_Q * 4, stream);
                const _Float16* WihD = WihH + (size_t)dir * GATES_Q * xld;
                const _Float16* WhhD = WhhH + (size_t)dir * GATES_Q * NHID_Q;
                const float*    bd   = bb + (size_t)dir * GATES_Q;

                for (int s = 0; s < L_Q; s++) {
                    int t = dir ? (L_Q - 1 - s) : s;
                    gemm_wmma_f16<<<gemm_grid, 256, 0, stream>>>(
                        Xin + (size_t)t * B_Q * xld, WihD, bd, nullptr, gates,
                        B_Q, GATES_Q, xld, 1);
                    gemm_wmma_f16<<<gemm_grid, 256, 0, stream>>>(
                        Hf16, WhhD, nullptr, gates, gates,
                        B_Q, GATES_Q, NHID_Q, 2);

                    _Float16* dest; int dld, doff, dod;
                    if (layer == 0) {
                        dest = X1 + (size_t)t * B_Q * IN1_Q; dld = IN1_Q;
                        doff = dir * NHID_Q; dod = 1;
                    } else {
                        dest = repb; dld = REPP_Q;
                        doff = dir * NHID_Q; dod = (t == pos) ? 1 : 0;
                    }
                    k_lstm_cell<<<(B_Q * NHID_Q) / 256, 256, 0, stream>>>(
                        gates, Cbuf, Hf16, dest, dld, doff, dod);
                }
            }
        }

        k_rep_extra<<<(B_Q * 32) / 256, 256, 0, stream>>>(keypos, repb, pos);

        dim3 g2(1, B_Q / 128);
        gemm_wmma_f16<<<g2, 256, 0, stream>>>(repb, mlpWh, mlpb, nullptr, rep100,
                                              B_Q, 100, REPP_Q, 1);

        k_u<<<(B_Q * 298 + 255) / 256, 256, 0, stream>>>(rep100, mlp1W, ubuf);
        k_cb<<<(B_Q + 255) / 256, 256, 0, stream>>>(rep100, mlp1b, cbb);
        k_score<<<(B_Q * K_Q + 255) / 256, 256, 0, stream>>>(ubuf, cbb, encT, repw,
                                                             delv, holdv, scoreb, step);
        k_act<<<(B_Q + 255) / 256, 256, 0, stream>>>(scoreb, repw, valid, st, len,
                                                     out_pis, out_act, out_st, out_len,
                                                     pos, step);
    }
}
